// Deformconv_12799002542774
// MI455X (gfx1250) — matrix operand order repaired and checked by reading, not compiled, not run on hardware
//
#include <hip/hip_runtime.h>
#include <math.h>

// ---------------------------------------------------------------------------
// DCNv2 (deformable conv) for MI455X / gfx1250: wave32, WMMA bf16, TDM.
//   B=4, C=128, O=128, H=W=160, 9 taps, GEMM K-dim = 9*128 = 1152.
// Pipeline:
//   1) x NCHW f32 -> NHWC bf16 (workspace)
//   2) pack weights into exact WMMA-B per-lane layout (bf16)
//   3) offset/mask conv: TDM (tensor_load_to_lds, LDS-padded to avoid bank
//      conflicts) stages a 3x18x128 x-slab per block; barrier-free WMMA K-loop
//   4) main GEMM: fused bilinear-sample A-tile (full tap per barrier pair),
//      8 waves share the A tile, WMMA f32<-bf16 accumulate
// ---------------------------------------------------------------------------

typedef __attribute__((ext_vector_type(16))) __bf16       v16bf;
typedef __attribute__((ext_vector_type(8)))  float        v8f;
typedef __attribute__((ext_vector_type(4)))  unsigned int v4u;
typedef __attribute__((ext_vector_type(8)))  int          v8i;
typedef __attribute__((ext_vector_type(4)))  int          v4i;

union BF16x16 { v16bf v; uint4 q[2]; };
union BFPair  { unsigned int u; __bf16 h[2]; };

#define B_   4
#define C_   128
#define O_   128
#define H_   160
#define W_   160
#define HW_  (H_*W_)
#define NKC  36            // 1152 / 32 K-chunks

// ---------------------------------------------------------------------------
// Prep 1: NCHW f32 -> NHWC bf16
// ---------------------------------------------------------------------------
__global__ __launch_bounds__(256) void k_to_nhwc(const float* __restrict__ x,
                                                 __bf16* __restrict__ xn) {
    size_t i = (size_t)blockIdx.x * 256 + threadIdx.x;
    if (i >= (size_t)B_ * H_ * W_ * C_) return;
    int c = (int)(i % C_);
    size_t bhw = i / C_;
    int w = (int)(bhw % W_);
    size_t bh = bhw / W_;
    int h = (int)(bh % H_);
    int b = (int)(bh / H_);
    xn[i] = (__bf16)x[(((size_t)b * C_ + c) * H_ + h) * W_ + w];
}

// ---------------------------------------------------------------------------
// Prep 2: main weights (O,C,3,3) -> packed bf16 [kc][n][half][16]
//   K index kk = tap*128 + c; lane<16 -> K 0..15 of chunk, lane>=16 -> 16..31
// ---------------------------------------------------------------------------
__global__ __launch_bounds__(256) void k_pack_w(const float* __restrict__ w,
                                                __bf16* __restrict__ wp) {
    int i = blockIdx.x * 256 + threadIdx.x;            // over 36*128*32
    if (i >= NKC * O_ * 32) return;
    int j = i & 15;
    int t = i >> 4;
    int half = t & 1;   t >>= 1;
    int n = t & 127;    int kc = t >> 7;
    int kk = kc * 32 + half * 16 + j;
    int tap = kk >> 7;
    int c   = kk & 127;
    wp[i] = (__bf16)w[((size_t)n * C_ + c) * 9 + tap];
}

// ---------------------------------------------------------------------------
// Prep 3: offset(18,C,3,3)+mask(9,C,3,3) -> packed bf16 [kc][n(32)][half][16]
// ---------------------------------------------------------------------------
__global__ __launch_bounds__(256) void k_pack_woff(const float* __restrict__ w_off,
                                                   const float* __restrict__ w_mask,
                                                   __bf16* __restrict__ wp) {
    int i = blockIdx.x * 256 + threadIdx.x;            // over 36*32*32
    if (i >= NKC * 32 * 32) return;
    int j = i & 15;
    int t = i >> 4;
    int half = t & 1;  t >>= 1;
    int n = t & 31;    int kc = t >> 5;
    int kk = kc * 32 + half * 16 + j;
    int tap = kk >> 7;
    int c   = kk & 127;
    float v = 0.f;
    if (n < 18)      v = w_off [((size_t)n        * C_ + c) * 9 + tap];
    else if (n < 27) v = w_mask[((size_t)(n - 18) * C_ + c) * 9 + tap];
    wp[i] = (__bf16)v;
}

// ---------------------------------------------------------------------------
// Stage A: offset+mask conv3x3, implicit GEMM with WMMA bf16 + TDM staging.
//   Block = 64 threads (2 waves). Tile: 16 pos (M) x 32 ch (N).
//   TDM loads a 3-row x 18-col x 128-ch bf16 slab of x into LDS once, with
//   pad_interval=5 (256B) / pad_amount=3 (16B) => 272B column stride so the
//   per-lane ds_load_b128 A-fragment reads hit 16 distinct banks.
//   K-loop (9 taps x 4 chunks) then runs with NO barriers / NO LDS writes.
// ---------------------------------------------------------------------------
#define SLABW   18
#define SLABCOL 68          // dwords per column: 64 data + 4 pad (272 B)

__global__ __launch_bounds__(64) void k_offmask(const __bf16* __restrict__ xn,
                                                const __bf16* __restrict__ wp,
                                                const float*  __restrict__ b_off,
                                                const float*  __restrict__ b_mask,
                                                float* __restrict__ off_ws,
                                                float* __restrict__ mask_ws) {
    __shared__ unsigned int slab[3 * SLABW * SLABCOL];   // ~14.7 KB

    const int tid  = threadIdx.x;                        // 0..63
    const int p0   = blockIdx.x * 16;
    const int b    = p0 / HW_;
    const int rem  = p0 % HW_;
    const int h    = rem / W_;
    const int w0   = rem % W_;

    const int lane = tid & 31;
    const int n0   = (tid >> 5) * 16;                    // wave -> N columns
    const int aM   = lane & 15;
    const int ah   = lane >> 4;

    // Edge blocks: pre-zero slab (rows/cols TDM skips stay zero). Uniform branch.
    const bool boundary = (h == 0) | (h == H_ - 1) | (w0 == 0) | (w0 + 16 == W_);
    if (boundary) {
        for (int i = tid; i < 3 * SLABW * SLABCOL; i += 64) slab[i] = 0u;
    }
    __syncthreads();

    // Wave 0 issues one TDM 1-D tile load per valid image row, then waits
    // TENSORcnt==0 before the workgroup barrier releases the readers.
    if (tid < 32) {
        // Flat shared-aperture address: low 32 bits are the LDS byte offset.
        const unsigned int ldsBase = (unsigned int)(size_t)(void*)slab;
        for (int r = 0; r < 3; ++r) {
            const int yy = h - 1 + r;
            if (yy < 0 || yy >= H_) continue;
            const int cl = max(w0 - 1, 0);
            const int cr = min(w0 + 16, W_ - 1);
            const unsigned int units = (unsigned int)(cr - cl + 1) * 32u; // 8B units
            const unsigned long long ga = (unsigned long long)(size_t)
                (xn + (((size_t)b * H_ + yy) * W_ + cl) * C_);
            const unsigned int lds = ldsBase +
                (unsigned int)((r * SLABW + (cl - (w0 - 1))) * SLABCOL) * 4u;
            // D# group 0: count=1 | lds_addr | global_addr[56:0] | type=2
            v4u g0 = { 1u, lds, (unsigned int)(ga & 0xffffffffu),
                       (unsigned int)((ga >> 32) & 0x1ffffffu) | (2u << 30) };
            // D# group 1: data_size=3(8B), pad_enable, pad_interval=5(256B),
            // pad_amount=3(16B); tensor_dim0=units, tensor_dim1=1,
            // tile_dim0=units, tile_dim1/2=0, dim0_stride=units.
            v8i g1 = { (int)((3u << 16) | (1u << 20) | (5u << 22) | (3u << 25)),
                       (int)((units & 0xffffu) << 16),
                       (int)((units >> 16) | (1u << 16)),
                       (int)(units << 16),
                       0,
                       (int)units,
                       0, 0 };
            v4i gz = { 0, 0, 0, 0 };
#if defined(__clang_major__) && (__clang_major__ >= 23)
            v8i gz8 = { 0, 0, 0, 0, 0, 0, 0, 0 };
            __builtin_amdgcn_tensor_load_to_lds(g0, g1, gz, gz, gz8, 0);
#else
            __builtin_amdgcn_tensor_load_to_lds(g0, g1, gz, gz, 0);
#endif
        }
        __builtin_amdgcn_s_wait_tensorcnt(0);
    }
    __syncthreads();

    v8f acc = {};
    for (int tap = 0; tap < 9; ++tap) {
        const int ky = tap / 3, kx = tap % 3;
        const unsigned int* colBase = slab + (ky * SLABW + aM + kx) * SLABCOL;
        #pragma unroll
        for (int cc = 0; cc < 4; ++cc) {
            BF16x16 Af, Bf;
            const unsigned int* ap = colBase + cc * 16 + ah * 8;
            Af.q[0] = *(const uint4*)ap;
            Af.q[1] = *(const uint4*)(ap + 4);
            const int kc = tap * 4 + cc;
            const uint4* wq = (const uint4*)(wp +
                              ((size_t)(kc * 32 + n0 + aM) * 2 + ah) * 16);
            Bf.q[0] = wq[0];
            Bf.q[1] = wq[1];
            acc = __builtin_amdgcn_wmma_f32_16x16x32_bf16(
                      false, Af.v, false, Bf.v, (short)0, acc, false, false);
        }
    }

    // Epilogue: D layout — VGPR r: lane<16 -> (M=r, N=lane); lane>=16 -> M=r+8
    const int N    = aM;
    const int mofs = ah * 8;
    const int n    = n0 + N;
    #pragma unroll
    for (int r = 0; r < 8; ++r) {
        const int hw = h * W_ + w0 + mofs + r;
        const float v = acc[r];
        if (n < 18) {
            off_ws[((size_t)b * 18 + n) * HW_ + hw] = v + b_off[n];
        } else if (n < 27) {
            float t = v + b_mask[n - 18];
            mask_ws[((size_t)b * 9 + (n - 18)) * HW_ + hw] = 1.f / (1.f + expf(-t));
        }
    }
}

// ---------------------------------------------------------------------------
// Stage B: main contraction out[m,o] = sum_{k,c} val[m, k*128+c] * w[o,c,k]
//   Block = 256 threads (8 waves). Tile: 16 positions x 128 out channels.
//   Per tap: each thread bilinear-samples one (pos, ch-pair) for all 4 chunks
//   (weights hoisted), writes the full 16x128 bf16 tap tile, one barrier pair,
//   then 4 barrier-free WMMAs per wave. A rows padded to 272 B (bank spread).
// ---------------------------------------------------------------------------
__global__ __launch_bounds__(256) void k_dcn_main(const __bf16* __restrict__ xn,
                                                  const float*  __restrict__ off_ws,
                                                  const float*  __restrict__ mask_ws,
                                                  const __bf16* __restrict__ wp,
                                                  const float*  __restrict__ bias,
                                                  float* __restrict__ out) {
    __shared__ unsigned int Atile[16 * 68];   // [pos][64 data + 4 pad] dwords

    const int tid  = threadIdx.x;             // 0..255
    const int p0   = blockIdx.x * 16;
    const int b    = p0 / HW_;
    const int rem  = p0 % HW_;
    const int h    = rem / W_;
    const int w0   = rem % W_;

    const int lane = tid & 31;
    const int n0   = (tid >> 5) * 16;         // wave id * 16 output channels
    const int aM   = lane & 15;
    const int ah   = lane >> 4;

    const int pos  = tid >> 4;                // sampling role: position 0..15
    const int cp   = tid & 15;                // channel pair 0..15

    const float* offB = off_ws  + (size_t)b * 18 * HW_ + (h * W_ + w0 + pos);
    const float* mB   = mask_ws + (size_t)b * 9  * HW_ + (h * W_ + w0 + pos);

    v8f acc = {};

    for (int tap = 0; tap < 9; ++tap) {
        // per-(pos,tap) bilinear setup, shared across the 4 channel chunks
        const float dy = offB[(size_t)(2 * tap)     * HW_];
        const float dx = offB[(size_t)(2 * tap + 1) * HW_];
        const float mk = mB[(size_t)tap * HW_];
        const float py = (float)(h + tap / 3 - 1) + dy;
        const float px = (float)(w0 + pos + tap % 3 - 1) + dx;
        const float y0f = floorf(py), x0f = floorf(px);
        const float ly = py - y0f, lx = px - x0f;
        const int y0 = (int)y0f, x0i = (int)x0f;
        const int y1 = y0 + 1,   x1  = x0i + 1;
        const bool vy0 = (y0 >= 0) & (y0 < H_);
        const bool vy1 = (y1 >= 0) & (y1 < H_);
        const bool vx0 = (x0i >= 0) & (x0i < W_);
        const bool vx1 = (x1 >= 0) & (x1 < W_);
        const float w00 = (1.f - ly) * (1.f - lx) * ((vy0 & vx0) ? mk : 0.f);
        const float w01 = (1.f - ly) * lx         * ((vy0 & vx1) ? mk : 0.f);
        const float w10 = ly * (1.f - lx)         * ((vy1 & vx0) ? mk : 0.f);
        const float w11 = ly * lx                 * ((vy1 & vx1) ? mk : 0.f);
        const int cy0 = min(max(y0, 0), H_ - 1);
        const int cy1 = min(max(y1, 0), H_ - 1);
        const int cx0 = min(max(x0i, 0), W_ - 1);
        const int cx1 = min(max(x1, 0), W_ - 1);
        const __bf16* s00 = xn + (((size_t)b * H_ + cy0) * W_ + cx0) * C_;
        const __bf16* s01 = xn + (((size_t)b * H_ + cy0) * W_ + cx1) * C_;
        const __bf16* s10 = xn + (((size_t)b * H_ + cy1) * W_ + cx0) * C_;
        const __bf16* s11 = xn + (((size_t)b * H_ + cy1) * W_ + cx1) * C_;

        unsigned int smp[4];
        #pragma unroll
        for (int cc = 0; cc < 4; ++cc) {
            const int c = cc * 32 + cp * 2;
            BFPair a0, a1, a2, a3, rr;
            a0.u = *(const unsigned int*)(s00 + c);
            a1.u = *(const unsigned int*)(s01 + c);
            a2.u = *(const unsigned int*)(s10 + c);
            a3.u = *(const unsigned int*)(s11 + c);
            const float v0 = w00 * (float)a0.h[0] + w01 * (float)a1.h[0]
                           + w10 * (float)a2.h[0] + w11 * (float)a3.h[0];
            const float v1 = w00 * (float)a0.h[1] + w01 * (float)a1.h[1]
                           + w10 * (float)a2.h[1] + w11 * (float)a3.h[1];
            rr.h[0] = (__bf16)v0;
            rr.h[1] = (__bf16)v1;
            smp[cc] = rr.u;
        }

        __syncthreads();                       // previous tap's reads complete
        #pragma unroll
        for (int cc = 0; cc < 4; ++cc) Atile[pos * 68 + cc * 16 + cp] = smp[cc];
        __syncthreads();                       // tap tile visible to all waves

        #pragma unroll
        for (int cc = 0; cc < 4; ++cc) {
            BF16x16 Af, Bf;
            const unsigned int* ap = Atile + aM * 68 + cc * 16 + ah * 8;
            Af.q[0] = *(const uint4*)ap;
            Af.q[1] = *(const uint4*)(ap + 4);
            const int kc = tap * 4 + cc;
            const uint4* wq = (const uint4*)(wp +
                              ((size_t)(kc * O_ + n0 + aM) * 2 + ah) * 16);
            Bf.q[0] = wq[0];
            Bf.q[1] = wq[1];
            acc = __builtin_amdgcn_wmma_f32_16x16x32_bf16(
                      false, Af.v, false, Bf.v, (short)0, acc, false, false);
        }
    }

    // Epilogue: out[b][oc][h][w0+M] = acc + bias[oc]
    const int N    = aM;
    const int mofs = ah * 8;
    const int oc   = n0 + N;
    const float bs = bias[oc];
    float* op = out + (((size_t)b * O_ + oc) * H_ + h) * W_ + w0 + mofs;
    #pragma unroll
    for (int r = 0; r < 8; ++r) op[r] = acc[r] + bs;
}

// ---------------------------------------------------------------------------
// Launch
// ---------------------------------------------------------------------------
extern "C" void kernel_launch(void* const* d_in, const int* in_sizes, int n_in,
                              void* d_out, int out_size, void* d_ws, size_t ws_size,
                              hipStream_t stream) {
    const float* x      = (const float*)d_in[0];
    const float* w_off  = (const float*)d_in[1];
    const float* b_off  = (const float*)d_in[2];
    const float* w_mask = (const float*)d_in[3];
    const float* b_mask = (const float*)d_in[4];
    const float* w      = (const float*)d_in[5];
    const float* bias   = (const float*)d_in[6];
    float* out = (float*)d_out;

    char* ws = (char*)d_ws;
    size_t o = 0;
    auto align256 = [](size_t v) { return (v + 255) & ~(size_t)255; };
    __bf16* xn      = (__bf16*)(ws + o); o = align256(o + (size_t)B_ * H_ * W_ * C_ * 2);
    float*  off_ws  = (float*)(ws + o);  o = align256(o + (size_t)B_ * 18 * HW_ * 4);
    float*  mask_ws = (float*)(ws + o);  o = align256(o + (size_t)B_ * 9  * HW_ * 4);
    __bf16* wpM     = (__bf16*)(ws + o); o = align256(o + (size_t)NKC * O_ * 32 * 2);
    __bf16* wpO     = (__bf16*)(ws + o); o = align256(o + (size_t)NKC * 32 * 32 * 2);
    (void)ws_size; (void)in_sizes; (void)n_in; (void)out_size;

    const int nElemX = B_ * H_ * W_ * C_;
    k_to_nhwc  <<<(nElemX + 255) / 256, 256, 0, stream>>>(x, xn);
    k_pack_w   <<<(NKC * O_ * 32 + 255) / 256, 256, 0, stream>>>(w, wpM);
    k_pack_woff<<<(NKC * 32 * 32 + 255) / 256, 256, 0, stream>>>(w_off, w_mask, wpO);

    const int nTiles = B_ * HW_ / 16;          // 6400
    k_offmask  <<<nTiles,  64, 0, stream>>>(xn, wpO, b_off, b_mask, off_ws, mask_ws);
    k_dcn_main <<<nTiles, 256, 0, stream>>>(xn, off_ws, mask_ws, wpM, bias, out);
}
